// STPSynapse_9182640079535
// MI455X (gfx1250) — compile-verified, matrix operand order checked by reading, not run on hardware
//
#include <hip/hip_runtime.h>

// Problem constants (match reference)
#define B_TOT   8192
#define IN_F    4096
#define OUT_F   128

// Tiling
#define KC            32                 // K per chunk == K of one bf16 WMMA
#define KSTR          40                 // padded LDS row stride (bf16 elems) -> 80B, conflict-free
#define ROWS_PER_WAVE 16
#define WAVES_PER_BLK 2
#define ROWS_PER_BLK  (ROWS_PER_WAVE * WAVES_PER_BLK)
#define THREADS       (32 * WAVES_PER_BLK)

typedef __bf16 v16bf __attribute__((ext_vector_type(16)));
typedef __bf16 v8bf  __attribute__((ext_vector_type(8)));
typedef __bf16 v4bf  __attribute__((ext_vector_type(4)));
typedef float  v8f   __attribute__((ext_vector_type(8)));
typedef float  v4f   __attribute__((ext_vector_type(4)));

union AF { v16bf v; v8bf h[2]; };

__device__ __forceinline__ void split_bf16(float f, __bf16& hi, __bf16& lo) {
    hi = (__bf16)f;                      // RNE truncation to bf16
    lo = (__bf16)(f - (float)hi);        // residual -> ~16 effective mantissa bits
}

// ---------------------------------------------------------------------------
// Prologue: pack weight once into bf16 hi/lo planes in workspace (2 MB total,
// L2-resident for the whole launch).
// ---------------------------------------------------------------------------
__global__ __launch_bounds__(256)
void pack_weight_kernel(const float* __restrict__ w,
                        __bf16* __restrict__ whi, __bf16* __restrict__ wlo)
{
    int t = blockIdx.x * 256 + threadIdx.x;       // one float4 per thread
    v4f w4 = *reinterpret_cast<const v4f*>(&w[(size_t)t * 4]);
    v4bf hv, lv;
    #pragma unroll
    for (int e = 0; e < 4; ++e) {
        __bf16 h, l; split_bf16(w4[e], h, l);
        hv[e] = h; lv[e] = l;
    }
    *(v4bf*)&whi[(size_t)t * 4] = hv;
    *(v4bf*)&wlo[(size_t)t * 4] = lv;
}

// ---------------------------------------------------------------------------
// Main fused kernel, v2: async global->LDS weight staging (gfx1250
// global_load_async_to_lds_b128), overlapped with the elementwise STP phase.
// ---------------------------------------------------------------------------
__global__ __launch_bounds__(THREADS)
void stp_fused_async_kernel(const float* __restrict__ pre,
                            const __bf16* __restrict__ whi,
                            const __bf16* __restrict__ wlo,
                            const float* __restrict__ u, const float* __restrict__ x,
                            float* __restrict__ out, float* __restrict__ u_out,
                            float* __restrict__ x_out)
{
    const float FAC = 0.990049833749168107f;  // exp(-1/100)
    const float DEP = 0.995012479192682309f;  // exp(-1/200)
    const float UC  = 0.5f;

    __shared__ __align__(16) __bf16 sWhi[OUT_F * KSTR];
    __shared__ __align__(16) __bf16 sWlo[OUT_F * KSTR];
    __shared__ __align__(16) __bf16 sAhi[WAVES_PER_BLK][ROWS_PER_WAVE * KSTR];
    __shared__ __align__(16) __bf16 sAlo[WAVES_PER_BLK][ROWS_PER_WAVE * KSTR];

    const int  tid  = threadIdx.x;
    const int  lane = tid & 31;
    const int  wv   = tid >> 5;
    const long rowBaseWave = (long)blockIdx.x * ROWS_PER_BLK + (long)wv * ROWS_PER_WAVE;

    v8f acc[8];
    #pragma unroll
    for (int i = 0; i < 8; ++i)
        #pragma unroll
        for (int r = 0; r < 8; ++r) acc[i][r] = 0.0f;

    for (int k0 = 0; k0 < IN_F; k0 += KC) {
        // ---- Async-stage weight tile [128 x KC] hi/lo -> padded LDS layout.
        // Each lane copies 16B segments; KC bf16 = 64B = 4 segments per row.
        // 512 segments per plane / 64 threads = 8 per thread. Overlaps with
        // the elementwise phase below; fenced by s_wait_asynccnt + barrier.
        #pragma unroll
        for (int j = 0; j < (OUT_F * KC * 2 / 16) / THREADS; ++j) {   // 8 iters
            int t   = j * THREADS + tid;
            int row = t >> 2;
            int seg = (t & 3) * 16;                      // byte offset in row
            unsigned lds_hi = (unsigned)(unsigned long long)&sWhi[row * KSTR] + seg;
            unsigned lds_lo = (unsigned)(unsigned long long)&sWlo[row * KSTR] + seg;
            unsigned long long ga_hi =
                (unsigned long long)&whi[(size_t)row * IN_F + k0] + seg;
            unsigned long long ga_lo =
                (unsigned long long)&wlo[(size_t)row * IN_F + k0] + seg;
            asm volatile("global_load_async_to_lds_b128 %0, %1, off"
                         :: "v"(lds_hi), "v"(ga_hi) : "memory");
            asm volatile("global_load_async_to_lds_b128 %0, %1, off"
                         :: "v"(lds_lo), "v"(ga_lo) : "memory");
        }

        // ---- Elementwise STP phase (per wave: 16 rows x KC cols, coalesced,
        // non-temporal: stream data must not churn L2) ----
        #pragma unroll
        for (int j = 0; j < (ROWS_PER_WAVE * KC / 4) / 32; ++j) { // 4 iters
            int  t   = j * 32 + lane;
            int  row = t >> 3;               // 0..15
            int  col = (t & 7) * 4;          // 0..28
            long g   = (rowBaseWave + row) * (long)IN_F + k0 + col;

            v4f s4 = __builtin_nontemporal_load(reinterpret_cast<const v4f*>(&pre[g]));
            v4f u4 = __builtin_nontemporal_load(reinterpret_cast<const v4f*>(&u[g]));
            v4f x4 = __builtin_nontemporal_load(reinterpret_cast<const v4f*>(&x[g]));

            v4f un4, xn4; v4bf ah, al;
            #pragma unroll
            for (int e = 0; e < 4; ++e) {
                float s = s4[e], uu = u4[e], xx = x4[e];
                float rel = uu * xx * s;                       // pre_spike * u * x
                float ufd = uu * FAC;
                float un  = ufd + UC * (1.0f - ufd) * s;       // u_new
                float xn  = xx * DEP + (1.0f - xx) * DEP * (1.0f - s * un); // x_new
                un4[e] = un; xn4[e] = xn;
                __bf16 h, l; split_bf16(rel, h, l);
                ah[e] = h; al[e] = l;
            }
            __builtin_nontemporal_store(un4, reinterpret_cast<v4f*>(&u_out[g]));
            __builtin_nontemporal_store(xn4, reinterpret_cast<v4f*>(&x_out[g]));
            *(v4bf*)&sAhi[wv][row * KSTR + col] = ah;
            *(v4bf*)&sAlo[wv][row * KSTR + col] = al;
        }

        // Fence this wave's async LDS writes, then sync the workgroup.
        asm volatile("s_wait_asynccnt 0x0" ::: "memory");
        __syncthreads();

        // ---- WMMA phase: 16-bit A/B fragment layout (ISA 7.12.2) ----
        const int ar   = lane & 15;
        const int koff = (lane >> 4) * 8;    // lanes 0-15: K0-7 & 16-23; 16-31: K8-15 & 24-31
        AF ahi, alo;
        ahi.h[0] = *(const v8bf*)&sAhi[wv][ar * KSTR + koff];
        ahi.h[1] = *(const v8bf*)&sAhi[wv][ar * KSTR + koff + 16];
        alo.h[0] = *(const v8bf*)&sAlo[wv][ar * KSTR + koff];
        alo.h[1] = *(const v8bf*)&sAlo[wv][ar * KSTR + koff + 16];

        #pragma unroll
        for (int nt = 0; nt < 8; ++nt) {
            int wrow = nt * 16 + ar;         // B[k][n] = weight[n][k]
            AF bhi, blo;
            bhi.h[0] = *(const v8bf*)&sWhi[wrow * KSTR + koff];
            bhi.h[1] = *(const v8bf*)&sWhi[wrow * KSTR + koff + 16];
            blo.h[0] = *(const v8bf*)&sWlo[wrow * KSTR + koff];
            blo.h[1] = *(const v8bf*)&sWlo[wrow * KSTR + koff + 16];
            // bf16x2 split product: hi*hi + hi*lo + lo*hi, f32 accumulate
            acc[nt] = __builtin_amdgcn_wmma_f32_16x16x32_bf16(false, ahi.v, false, bhi.v,
                                                              (short)0, acc[nt], false, false);
            acc[nt] = __builtin_amdgcn_wmma_f32_16x16x32_bf16(false, ahi.v, false, blo.v,
                                                              (short)0, acc[nt], false, false);
            acc[nt] = __builtin_amdgcn_wmma_f32_16x16x32_bf16(false, alo.v, false, bhi.v,
                                                              (short)0, acc[nt], false, false);
        }
        __syncthreads();
    }

    // ---- Epilogue: 32-bit C/D layout -> out[b, o] ----
    const int n  = lane & 15;
    const int mo = (lane >> 4) * 8;          // VGPR r holds M = mo + r
    #pragma unroll
    for (int nt = 0; nt < 8; ++nt)
        #pragma unroll
        for (int r = 0; r < 8; ++r)
            __builtin_nontemporal_store(acc[nt][r],
                &out[(rowBaseWave + mo + r) * OUT_F + nt * 16 + n]);
}

// ---------------------------------------------------------------------------
// Fallback (round-1 version): self-contained, converts weights per chunk.
// Used only if the workspace is too small for the packed weights.
// ---------------------------------------------------------------------------
__global__ __launch_bounds__(THREADS)
void stp_fused_kernel(const float* __restrict__ pre, const float* __restrict__ w,
                      const float* __restrict__ u,   const float* __restrict__ x,
                      float* __restrict__ out, float* __restrict__ u_out,
                      float* __restrict__ x_out)
{
    const float FAC = 0.990049833749168107f;
    const float DEP = 0.995012479192682309f;
    const float UC  = 0.5f;

    __shared__ __align__(16) __bf16 sWhi[OUT_F * KSTR];
    __shared__ __align__(16) __bf16 sWlo[OUT_F * KSTR];
    __shared__ __align__(16) __bf16 sAhi[WAVES_PER_BLK][ROWS_PER_WAVE * KSTR];
    __shared__ __align__(16) __bf16 sAlo[WAVES_PER_BLK][ROWS_PER_WAVE * KSTR];

    const int  tid  = threadIdx.x;
    const int  lane = tid & 31;
    const int  wv   = tid >> 5;
    const long rowBaseWave = (long)blockIdx.x * ROWS_PER_BLK + (long)wv * ROWS_PER_WAVE;

    v8f acc[8];
    #pragma unroll
    for (int i = 0; i < 8; ++i)
        #pragma unroll
        for (int r = 0; r < 8; ++r) acc[i][r] = 0.0f;

    for (int k0 = 0; k0 < IN_F; k0 += KC) {
        #pragma unroll
        for (int j = 0; j < (OUT_F * KC / 4) / THREADS; ++j) {
            int t   = j * THREADS + tid;
            int row = t >> 3;
            int col = (t & 7) * 4;
            v4f w4  = *reinterpret_cast<const v4f*>(&w[(long)row * IN_F + k0 + col]);
            v4bf hv, lv;
            #pragma unroll
            for (int e = 0; e < 4; ++e) {
                __bf16 h, l; split_bf16(w4[e], h, l);
                hv[e] = h; lv[e] = l;
            }
            *(v4bf*)&sWhi[row * KSTR + col] = hv;
            *(v4bf*)&sWlo[row * KSTR + col] = lv;
        }
        #pragma unroll
        for (int j = 0; j < (ROWS_PER_WAVE * KC / 4) / 32; ++j) {
            int  t   = j * 32 + lane;
            int  row = t >> 3;
            int  col = (t & 7) * 4;
            long g   = (rowBaseWave + row) * (long)IN_F + k0 + col;
            v4f s4 = *reinterpret_cast<const v4f*>(&pre[g]);
            v4f u4 = *reinterpret_cast<const v4f*>(&u[g]);
            v4f x4 = *reinterpret_cast<const v4f*>(&x[g]);
            v4f un4, xn4; v4bf ah, al;
            #pragma unroll
            for (int e = 0; e < 4; ++e) {
                float s = s4[e], uu = u4[e], xx = x4[e];
                float rel = uu * xx * s;
                float ufd = uu * FAC;
                float un  = ufd + UC * (1.0f - ufd) * s;
                float xn  = xx * DEP + (1.0f - xx) * DEP * (1.0f - s * un);
                un4[e] = un; xn4[e] = xn;
                __bf16 h, l; split_bf16(rel, h, l);
                ah[e] = h; al[e] = l;
            }
            *reinterpret_cast<v4f*>(&u_out[g]) = un4;
            *reinterpret_cast<v4f*>(&x_out[g]) = xn4;
            *(v4bf*)&sAhi[wv][row * KSTR + col] = ah;
            *(v4bf*)&sAlo[wv][row * KSTR + col] = al;
        }
        __syncthreads();

        const int ar   = lane & 15;
        const int koff = (lane >> 4) * 8;
        AF ahi, alo;
        ahi.h[0] = *(const v8bf*)&sAhi[wv][ar * KSTR + koff];
        ahi.h[1] = *(const v8bf*)&sAhi[wv][ar * KSTR + koff + 16];
        alo.h[0] = *(const v8bf*)&sAlo[wv][ar * KSTR + koff];
        alo.h[1] = *(const v8bf*)&sAlo[wv][ar * KSTR + koff + 16];
        #pragma unroll
        for (int nt = 0; nt < 8; ++nt) {
            int wrow = nt * 16 + ar;
            AF bhi, blo;
            bhi.h[0] = *(const v8bf*)&sWhi[wrow * KSTR + koff];
            bhi.h[1] = *(const v8bf*)&sWhi[wrow * KSTR + koff + 16];
            blo.h[0] = *(const v8bf*)&sWlo[wrow * KSTR + koff];
            blo.h[1] = *(const v8bf*)&sWlo[wrow * KSTR + koff + 16];
            acc[nt] = __builtin_amdgcn_wmma_f32_16x16x32_bf16(false, ahi.v, false, bhi.v,
                                                              (short)0, acc[nt], false, false);
            acc[nt] = __builtin_amdgcn_wmma_f32_16x16x32_bf16(false, ahi.v, false, blo.v,
                                                              (short)0, acc[nt], false, false);
            acc[nt] = __builtin_amdgcn_wmma_f32_16x16x32_bf16(false, alo.v, false, bhi.v,
                                                              (short)0, acc[nt], false, false);
        }
        __syncthreads();
    }

    const int n  = lane & 15;
    const int mo = (lane >> 4) * 8;
    #pragma unroll
    for (int nt = 0; nt < 8; ++nt)
        #pragma unroll
        for (int r = 0; r < 8; ++r)
            out[(rowBaseWave + mo + r) * OUT_F + nt * 16 + n] = acc[nt][r];
}

extern "C" void kernel_launch(void* const* d_in, const int* in_sizes, int n_in,
                              void* d_out, int out_size, void* d_ws, size_t ws_size,
                              hipStream_t stream) {
    const float* pre = (const float*)d_in[0];
    const float* w   = (const float*)d_in[1];
    const float* u   = (const float*)d_in[2];
    const float* x   = (const float*)d_in[3];

    float* out = (float*)d_out;                              // [8192,128]
    float* uo  = out + (size_t)B_TOT * OUT_F;                // [8192,4096]
    float* xo  = uo  + (size_t)B_TOT * IN_F;                 // [8192,4096]

    dim3 grid(B_TOT / ROWS_PER_BLK);   // 256 blocks
    dim3 block(THREADS);               // 64 threads = 2 wave32

    const size_t planeElems = (size_t)OUT_F * IN_F;          // 524288
    const size_t needWs     = planeElems * 2 * sizeof(__bf16); // 2 MB

    if (ws_size >= needWs) {
        __bf16* whi = (__bf16*)d_ws;
        __bf16* wlo = whi + planeElems;
        pack_weight_kernel<<<dim3((unsigned)(planeElems / 4 / 256)), dim3(256), 0, stream>>>(
            w, whi, wlo);
        stp_fused_async_kernel<<<grid, block, 0, stream>>>(pre, whi, wlo, u, x, out, uo, xo);
    } else {
        stp_fused_kernel<<<grid, block, 0, stream>>>(pre, w, u, x, out, uo, xo);
    }
}
